// CausalSelfAttention_34093450395927
// MI455X (gfx1250) — matrix operand order repaired and checked by reading, not compiled, not run on hardware
//
#include <hip/hip_runtime.h>
#include <hip/hip_bf16.h>

// ---------------------------------------------------------------------------
// Causal self-attention for MI455X (gfx1250, wave32, WMMA).
// All GEMMs run through v_wmma_f32_16x16x32_f16 (f16 in, f32 accumulate).
// GEMM tiles are staged with GLOBAL_LOAD_ASYNC_TO_LDS_B128 (ASYNCcnt) with
// explicit double-buffering for copy/compute overlap; falls back to b128
// load/store staging if the builtin is unavailable.
// ---------------------------------------------------------------------------

typedef __attribute__((ext_vector_type(16))) _Float16 v16h;
typedef __attribute__((ext_vector_type(8)))  _Float16 v8h;
typedef __attribute__((ext_vector_type(8)))  float    v8f;
typedef __attribute__((ext_vector_type(4)))  int      v4i;

union Frag16 { v16h v; v8h h[2]; };

#define DMODEL 1024
#define NHEAD  16
#define DHEAD  64
#define SEQ    2048
#define BATCH  2
#define MTOT   (BATCH * SEQ)   // 4096

#if defined(__AMDGCN__) && __has_builtin(__builtin_amdgcn_global_load_async_to_lds_b128)
#define HAVE_ASYNC_LDS 1
#else
#define HAVE_ASYNC_LDS 0
#endif

#define AS_GLOBAL __attribute__((address_space(1)))
#define AS_LDS    __attribute__((address_space(3)))

// Stage 32 bytes (16 halves): global -> LDS.  Async path uses the CDNA5
// async-to-LDS DMA (no staging VGPRs); fallback is a b128 copy pair.
static __device__ __forceinline__ void stage_row(const _Float16* g, _Float16* l) {
#if HAVE_ASYNC_LDS
  __builtin_amdgcn_global_load_async_to_lds_b128(
      (AS_GLOBAL v4i*)g, (AS_LDS v4i*)l, 0, 0);
  __builtin_amdgcn_global_load_async_to_lds_b128(
      (AS_GLOBAL v4i*)(g + 8), (AS_LDS v4i*)(l + 8), 0, 0);
#else
  *(v8h*)l       = *(const v8h*)g;
  *(v8h*)(l + 8) = *(const v8h*)(g + 8);
#endif
}

static __device__ __forceinline__ void wait_stage() {
#if HAVE_ASYNC_LDS
#if __has_builtin(__builtin_amdgcn_s_wait_asynccnt)
  __builtin_amdgcn_s_wait_asynccnt(0);
#else
  asm volatile("s_wait_asynccnt 0" ::: "memory");
#endif
#endif
}

static __device__ __forceinline__ v8f wmma_f16(const Frag16& a, const Frag16& b, v8f c) {
  return __builtin_amdgcn_wmma_f32_16x16x32_f16(false, a.v, false, b.v, (short)0, c,
                                                false, false);
}

// ---------------------------------------------------------------------------
// Kernel 0a: fp32 -> fp16 conversion (x activations)
// ---------------------------------------------------------------------------
__global__ void cvt_f32_f16(const float* __restrict__ src, _Float16* __restrict__ dst, int n) {
  int i = blockIdx.x * blockDim.x + threadIdx.x;
  int stride = gridDim.x * blockDim.x;
  for (; i < n; i += stride) dst[i] = (_Float16)src[i];
}

// ---------------------------------------------------------------------------
// Kernel 0b: fp32 -> fp16 conversion + transpose.  src[K][N] -> dst[N][K].
// ---------------------------------------------------------------------------
__launch_bounds__(256)
__global__ void cvt_t_f32_f16(const float* __restrict__ src, _Float16* __restrict__ dst,
                              int K, int N) {
  __shared__ _Float16 tile[32][33];
  const int tid = threadIdx.x;
  const int nb  = blockIdx.x * 32;
  const int kb  = blockIdx.y * 32;
  const int c   = tid & 31;
  const int r0  = tid >> 5;
#pragma unroll
  for (int p = 0; p < 4; ++p) {
    int r = r0 + p * 8;             // k offset
    tile[r][c] = (_Float16)src[(size_t)(kb + r) * N + nb + c];
  }
  __syncthreads();
#pragma unroll
  for (int p = 0; p < 4; ++p) {
    int r = r0 + p * 8;             // n offset
    dst[(size_t)(nb + r) * K + kb + c] = tile[c][r];
  }
}

// ---------------------------------------------------------------------------
// Kernel 1: QKV projection.  X[4096,1024] x Wt[3072,1024] -> Q/K/V (f16)
//   Q -> [bh, t, d]   K -> [bh, t, d]   V -> transposed [bh, d, t]
// Block = 256 threads (8 waves), tile 128x128, K-step 32, double-buffered.
// ---------------------------------------------------------------------------
__launch_bounds__(256)
__global__ void qkv_gemm(const _Float16* __restrict__ X,
                         const _Float16* __restrict__ Wt,   // [3072][1024]
                         _Float16* __restrict__ Qh,
                         _Float16* __restrict__ Kh,
                         _Float16* __restrict__ Vt) {
  __shared__ _Float16 ldsA[2][128 * 40];
  __shared__ _Float16 ldsB[2][128 * 40];

  const int tid = threadIdx.x;
  const int wid = tid >> 5;
  const int lid = tid & 31;
  const int hi  = lid >> 4;
  const int l15 = lid & 15;
  const int m0  = blockIdx.x * 128;
  const int n0  = blockIdx.y * 128;
  const int wm  = wid & 3;      // 4 waves along M (32 rows each)
  const int wn  = wid >> 2;     // 2 waves along N (64 cols each)

  const int r    = tid >> 1;          // 0..127
  const int ko   = (tid & 1) * 16;    // 0 or 16
  const int ldso = r * 40 + ko;

  const _Float16* gA = X  + (size_t)(m0 + r) * DMODEL + ko;
  const _Float16* gB = Wt + (size_t)(n0 + r) * DMODEL + ko;

  v8f acc[2][4];
  const v8f vzero = {0.f, 0.f, 0.f, 0.f, 0.f, 0.f, 0.f, 0.f};
#pragma unroll
  for (int i = 0; i < 2; ++i)
#pragma unroll
    for (int j = 0; j < 4; ++j) acc[i][j] = vzero;

  // prologue: stage k-step 0 into buffer 0
  stage_row(gA, &ldsA[0][ldso]);
  stage_row(gB, &ldsB[0][ldso]);
  wait_stage();
  __syncthreads();

  for (int kk = 0; kk < DMODEL; kk += 32) {
    const int cur = (kk >> 5) & 1;
    const int nxt = cur ^ 1;
    if (kk + 32 < DMODEL) {           // stage next tile while computing
      stage_row(gA + kk + 32, &ldsA[nxt][ldso]);
      stage_row(gB + kk + 32, &ldsB[nxt][ldso]);
    }

    Frag16 a[2], b[4];
#pragma unroll
    for (int mt = 0; mt < 2; ++mt) {
      const _Float16* p = &ldsA[cur][(wm * 32 + mt * 16 + l15) * 40 + hi * 8];
      a[mt].h[0] = *(const v8h*)p;          // K = 8*hi .. +7
      a[mt].h[1] = *(const v8h*)(p + 16);   // K = 16+8*hi .. +7
    }
#pragma unroll
    for (int nt = 0; nt < 4; ++nt) {
      const _Float16* p = &ldsB[cur][(wn * 64 + nt * 16 + l15) * 40 + hi * 8];
      b[nt].h[0] = *(const v8h*)p;          // K = 8*hi .. +7
      b[nt].h[1] = *(const v8h*)(p + 16);   // K = 16+8*hi .. +7
    }
#pragma unroll
    for (int mt = 0; mt < 2; ++mt)
#pragma unroll
      for (int nt = 0; nt < 4; ++nt)
        acc[mt][nt] = wmma_f16(a[mt], b[nt], acc[mt][nt]);

    wait_stage();       // next tile landed in LDS (async path)
    __syncthreads();    // all waves done with cur, nxt visible
  }

  // Epilogue: scatter to Q / K / Vt in f16
#pragma unroll
  for (int mt = 0; mt < 2; ++mt) {
#pragma unroll
    for (int nt = 0; nt < 4; ++nt) {
#pragma unroll
      for (int v = 0; v < 8; ++v) {
        int row = m0 + wm * 32 + mt * 16 + v + 8 * hi;   // 0..4095
        int col = n0 + wn * 64 + nt * 16 + l15;          // 0..3071
        int sec = col >> 10;
        int c   = col & 1023;
        int hh  = c >> 6, d = c & 63;
        int bb  = row >> 11, t = row & 2047;
        int bh  = bb * NHEAD + hh;
        _Float16 hv = (_Float16)acc[mt][nt][v];
        if (sec == 0)      Qh[((size_t)bh * SEQ + t) * DHEAD + d] = hv;
        else if (sec == 1) Kh[((size_t)bh * SEQ + t) * DHEAD + d] = hv;
        else               Vt[((size_t)bh * DHEAD + d) * SEQ + t] = hv;
      }
    }
  }
}

// ---------------------------------------------------------------------------
// Kernel 2: flash attention (causal, online softmax).
// One wave = 16 query rows; 8 waves/block cover 128 rows; Bc = 32 keys/step.
// Wave-uniform loop bounds keep EXEC all-ones (WMMA requirement).
// ---------------------------------------------------------------------------
__launch_bounds__(256)
__global__ void attn_kernel(const _Float16* __restrict__ Qh,
                            const _Float16* __restrict__ Kh,
                            const _Float16* __restrict__ Vt,
                            _Float16* __restrict__ Yh) {   // [4096, 1024]
  __shared__ _Float16 ldsP[8 * 16 * 40];   // per-wave 16x32 P scratch (padded)

  const int tid  = threadIdx.x;
  const int wid  = tid >> 5;
  const int lid  = tid & 31;
  const int hi   = lid >> 4;
  const int l15  = lid & 15;
  const int nqb  = SEQ / 128;                 // 16 q-blocks per (b,h)
  const int bh   = blockIdx.x / nqb;          // 0..31
  const int qblk = blockIdx.x % nqb;
  const int b    = bh >> 4;
  const int hh   = bh & 15;
  const int q0   = qblk * 128 + wid * 16;     // wave's query base

  const _Float16* Qp = Qh + (size_t)bh * SEQ * DHEAD;
  const _Float16* Kp = Kh + (size_t)bh * SEQ * DHEAD;
  const _Float16* Vp = Vt + (size_t)bh * DHEAD * SEQ;
  _Float16* lp = &ldsP[wid * 16 * 40];

  // Q fragments (16x64 = two 16x32 A-frags), loaded once
  Frag16 aq[2];
#pragma unroll
  for (int kd = 0; kd < 2; ++kd) {
    const _Float16* p = Qp + (size_t)(q0 + l15) * DHEAD + kd * 32 + hi * 8;
    aq[kd].h[0] = *(const v8h*)p;
    aq[kd].h[1] = *(const v8h*)(p + 16);
  }

  const v8f vzero = {0.f, 0.f, 0.f, 0.f, 0.f, 0.f, 0.f, 0.f};
  v8f o[4];
#pragma unroll
  for (int dt = 0; dt < 4; ++dt) o[dt] = vzero;
  float mrow[8], lrow[8];
#pragma unroll
  for (int v = 0; v < 8; ++v) { mrow[v] = -1e30f; lrow[v] = 0.f; }

  const float scale = 0.125f;   // 1/sqrt(64)

  for (int j0 = 0; j0 < q0 + 16; j0 += 32) {
    // ---- S = Q K^T  (two 16x16 n-tiles, two K-steps of 32 over d) ----
    v8f s[2];
#pragma unroll
    for (int nt = 0; nt < 2; ++nt) {
      Frag16 bk0, bk1;
      const _Float16* p0 = Kp + (size_t)(j0 + nt * 16 + l15) * DHEAD + hi * 8;
      bk0.h[0] = *(const v8h*)p0;       bk0.h[1] = *(const v8h*)(p0 + 16);
      const _Float16* p1 = p0 + 32;
      bk1.h[0] = *(const v8h*)p1;       bk1.h[1] = *(const v8h*)(p1 + 16);
      v8f z = vzero;
      z = wmma_f16(aq[0], bk0, z);
      z = wmma_f16(aq[1], bk1, z);
      s[nt] = z;
    }

    // ---- causal mask + online softmax (row = v + 8*hi, col = lane&15) ----
    float p0v[8], p1v[8], alpha[8];
#pragma unroll
    for (int v = 0; v < 8; ++v) {
      int qi  = q0 + v + 8 * hi;
      int k0c = j0 + l15;
      int k1c = j0 + 16 + l15;
      float s0 = (k0c <= qi) ? s[0][v] * scale : -1e30f;
      float s1 = (k1c <= qi) ? s[1][v] * scale : -1e30f;
      float r = fmaxf(s0, s1);
      r = fmaxf(r, __shfl_xor(r, 1));
      r = fmaxf(r, __shfl_xor(r, 2));
      r = fmaxf(r, __shfl_xor(r, 4));
      r = fmaxf(r, __shfl_xor(r, 8));     // stays within 16-lane half-group
      float mnew = fmaxf(mrow[v], r);
      float al = __expf(mrow[v] - mnew);
      float e0 = __expf(s0 - mnew);
      float e1 = __expf(s1 - mnew);
      float rs = e0 + e1;
      rs += __shfl_xor(rs, 1);
      rs += __shfl_xor(rs, 2);
      rs += __shfl_xor(rs, 4);
      rs += __shfl_xor(rs, 8);
      mrow[v] = mnew;
      lrow[v] = lrow[v] * al + rs;
      alpha[v] = al;
      p0v[v] = e0;
      p1v[v] = e1;
    }
#pragma unroll
    for (int dt = 0; dt < 4; ++dt)
#pragma unroll
      for (int v = 0; v < 8; ++v) o[dt][v] *= alpha[v];

    // ---- D-layout P -> LDS -> A-layout fragment ----
#pragma unroll
    for (int v = 0; v < 8; ++v) {
      lp[(v + 8 * hi) * 40 + l15]      = (_Float16)p0v[v];
      lp[(v + 8 * hi) * 40 + 16 + l15] = (_Float16)p1v[v];
    }
    asm volatile("s_wait_dscnt 0" ::: "memory");   // same-wave LDS RAW fence

    Frag16 ap;
    const _Float16* pp = lp + l15 * 40 + hi * 8;
    ap.h[0] = *(const v8h*)pp;
    ap.h[1] = *(const v8h*)(pp + 16);

    // ---- O += P * V  (V stored transposed [d][t]: contiguous B-frags) ----
#pragma unroll
    for (int dt = 0; dt < 4; ++dt) {
      Frag16 bv;
      const _Float16* pv = Vp + (size_t)(dt * 16 + l15) * SEQ + j0 + hi * 8;
      bv.h[0] = *(const v8h*)pv;
      bv.h[1] = *(const v8h*)(pv + 16);
      o[dt] = wmma_f16(ap, bv, o[dt]);
    }
  }

  // ---- normalize and write Y (f16) as [b*T+t][h*64+d] ----
#pragma unroll
  for (int v = 0; v < 8; ++v) {
    float inv = 1.0f / lrow[v];
    int row = b * SEQ + q0 + v + 8 * hi;
#pragma unroll
    for (int dt = 0; dt < 4; ++dt)
      Yh[(size_t)row * DMODEL + hh * DHEAD + dt * 16 + l15] =
          (_Float16)(o[dt][v] * inv);
  }
}

// ---------------------------------------------------------------------------
// Kernel 3: output projection.  Y[4096,1024] x Wpt[1024,1024] -> out fp32
// ---------------------------------------------------------------------------
__launch_bounds__(256)
__global__ void proj_gemm(const _Float16* __restrict__ Y,
                          const _Float16* __restrict__ Wt,
                          float* __restrict__ Out) {
  __shared__ _Float16 ldsA[2][128 * 40];
  __shared__ _Float16 ldsB[2][128 * 40];

  const int tid = threadIdx.x;
  const int wid = tid >> 5;
  const int lid = tid & 31;
  const int hi  = lid >> 4;
  const int l15 = lid & 15;
  const int m0  = blockIdx.x * 128;
  const int n0  = blockIdx.y * 128;
  const int wm  = wid & 3;
  const int wn  = wid >> 2;

  const int r    = tid >> 1;
  const int ko   = (tid & 1) * 16;
  const int ldso = r * 40 + ko;

  const _Float16* gA = Y  + (size_t)(m0 + r) * DMODEL + ko;
  const _Float16* gB = Wt + (size_t)(n0 + r) * DMODEL + ko;

  v8f acc[2][4];
  const v8f vzero = {0.f, 0.f, 0.f, 0.f, 0.f, 0.f, 0.f, 0.f};
#pragma unroll
  for (int i = 0; i < 2; ++i)
#pragma unroll
    for (int j = 0; j < 4; ++j) acc[i][j] = vzero;

  stage_row(gA, &ldsA[0][ldso]);
  stage_row(gB, &ldsB[0][ldso]);
  wait_stage();
  __syncthreads();

  for (int kk = 0; kk < DMODEL; kk += 32) {
    const int cur = (kk >> 5) & 1;
    const int nxt = cur ^ 1;
    if (kk + 32 < DMODEL) {
      stage_row(gA + kk + 32, &ldsA[nxt][ldso]);
      stage_row(gB + kk + 32, &ldsB[nxt][ldso]);
    }

    Frag16 a[2], b[4];
#pragma unroll
    for (int mt = 0; mt < 2; ++mt) {
      const _Float16* p = &ldsA[cur][(wm * 32 + mt * 16 + l15) * 40 + hi * 8];
      a[mt].h[0] = *(const v8h*)p;
      a[mt].h[1] = *(const v8h*)(p + 16);
    }
#pragma unroll
    for (int nt = 0; nt < 4; ++nt) {
      const _Float16* p = &ldsB[cur][(wn * 64 + nt * 16 + l15) * 40 + hi * 8];
      b[nt].h[0] = *(const v8h*)p;
      b[nt].h[1] = *(const v8h*)(p + 16);
    }
#pragma unroll
    for (int mt = 0; mt < 2; ++mt)
#pragma unroll
      for (int nt = 0; nt < 4; ++nt)
        acc[mt][nt] = wmma_f16(a[mt], b[nt], acc[mt][nt]);

    wait_stage();
    __syncthreads();
  }

#pragma unroll
  for (int mt = 0; mt < 2; ++mt)
#pragma unroll
    for (int nt = 0; nt < 4; ++nt)
#pragma unroll
      for (int v = 0; v < 8; ++v) {
        int row = m0 + wm * 32 + mt * 16 + v + 8 * hi;
        int col = n0 + wn * 64 + nt * 16 + l15;
        Out[(size_t)row * DMODEL + col] = acc[mt][nt][v];
      }
}

// ---------------------------------------------------------------------------
// Host launcher
// ---------------------------------------------------------------------------
extern "C" void kernel_launch(void* const* d_in, const int* in_sizes, int n_in,
                              void* d_out, int out_size, void* d_ws, size_t ws_size,
                              hipStream_t stream) {
  const float* x      = (const float*)d_in[0];   // [2,2048,1024]
  const float* w_attn = (const float*)d_in[1];   // [1024,3072]
  const float* w_proj = (const float*)d_in[2];   // [1024,1024]
  float* out = (float*)d_out;                    // [2,2048,1024]

  // Workspace carve-up (all f16, every segment a multiple of 2KB)
  _Float16* xh   = (_Float16*)d_ws;                      // 4096*1024
  _Float16* waht = xh   + (size_t)MTOT * DMODEL;         // Wt_attn [3072][1024]
  _Float16* wpht = waht + (size_t)DMODEL * 3 * DMODEL;   // Wt_proj [1024][1024]
  _Float16* qh   = wpht + (size_t)DMODEL * DMODEL;       // 32*2048*64
  _Float16* kh   = qh   + (size_t)BATCH * NHEAD * SEQ * DHEAD;
  _Float16* vt   = kh   + (size_t)BATCH * NHEAD * SEQ * DHEAD;
  _Float16* yh   = vt   + (size_t)BATCH * NHEAD * SEQ * DHEAD;  // 4096*1024

  cvt_f32_f16<<<2048, 256, 0, stream>>>(x, xh, MTOT * DMODEL);
  cvt_t_f32_f16<<<dim3((3 * DMODEL) / 32, DMODEL / 32), 256, 0, stream>>>(
      w_attn, waht, DMODEL, 3 * DMODEL);
  cvt_t_f32_f16<<<dim3(DMODEL / 32, DMODEL / 32), 256, 0, stream>>>(
      w_proj, wpht, DMODEL, DMODEL);

  qkv_gemm<<<dim3(MTOT / 128, (3 * DMODEL) / 128), 256, 0, stream>>>(xh, waht, qh, kh, vt);
  attn_kernel<<<BATCH * NHEAD * (SEQ / 128), 256, 0, stream>>>(qh, kh, vt, yh);
  proj_gemm<<<dim3(MTOT / 128, DMODEL / 128), 256, 0, stream>>>(yh, wpht, out);
}
